// MoE_26010321945291
// MI455X (gfx1250) — matrix operand order repaired and checked by reading, not compiled, not run on hardware
//
#include <hip/hip_runtime.h>
#include <hip/hip_bf16.h>
#include <math.h>

#define D_    1024
#define H_    4096
#define E_    8
#define NTOK  4096              // B*T = 2*2048
#define HROWS (NTOK + 128)      // hbuf row padding for straddling tiles

typedef __attribute__((ext_vector_type(16))) __bf16 v16bf;
typedef __attribute__((ext_vector_type(8)))  __bf16 v8bf;
typedef __attribute__((ext_vector_type(2)))  __bf16 v2bf;
typedef __attribute__((ext_vector_type(8)))  float  v8f;
typedef int v4i __attribute__((vector_size(16)));   // matches builtin param type

static __device__ __forceinline__ float fast_rcp(float x) {
#if __has_builtin(__builtin_amdgcn_rcpf)
    return __builtin_amdgcn_rcpf(x);
#else
    return 1.0f / x;
#endif
}

static __device__ __forceinline__ float softplus_f(float x) {
    return fmaxf(x, 0.0f) + log1pf(__expf(-fabsf(x)));
}

// async copy of 32B (two b128) from global bf16 row into LDS, else sync copy
static __device__ __forceinline__ void copy32_to_lds(const __bf16* gsrc, __bf16* ldst) {
#if defined(__AMDGCN__) && __has_builtin(__builtin_amdgcn_global_load_async_to_lds_b128) && \
    __has_builtin(__builtin_amdgcn_s_wait_asynccnt)
    typedef __attribute__((address_space(3))) v4i lv4i;
    __builtin_amdgcn_global_load_async_to_lds_b128((v4i*)gsrc, (lv4i*)ldst, 0, 0);
    __builtin_amdgcn_global_load_async_to_lds_b128((v4i*)gsrc, (lv4i*)ldst, 16, 0);
#else
    const uint4* s = (const uint4*)gsrc;
    uint4* d = (uint4*)ldst;
    d[0] = s[0];
    d[1] = s[1];
#endif
}

static __device__ __forceinline__ void wait_lds_copy() {
#if defined(__AMDGCN__) && __has_builtin(__builtin_amdgcn_global_load_async_to_lds_b128) && \
    __has_builtin(__builtin_amdgcn_s_wait_asynccnt)
    __builtin_amdgcn_s_wait_asynccnt(0);
#endif
}

// ---------------------------------------------------------------- convert x
__global__ void k_cvt_bf16(const float* __restrict__ src,
                           __bf16* __restrict__ dst, int n) {
    int i = (blockIdx.x * blockDim.x + threadIdx.x) * 4;
    if (i < n) {
        float4 v = *(const float4*)(src + i);
        dst[i + 0] = (__bf16)v.x;
        dst[i + 1] = (__bf16)v.y;
        dst[i + 2] = (__bf16)v.z;
        dst[i + 3] = (__bf16)v.w;
    }
}

// ---------------------------------------------------------------- router
__global__ void k_router(const float* __restrict__ x,
                         const float* __restrict__ noise,
                         const float* __restrict__ Wr, const float* __restrict__ br,
                         const float* __restrict__ Wn, const float* __restrict__ bn,
                         int2* __restrict__ topidx, float2* __restrict__ topgate) {
    int t = blockIdx.x * blockDim.x + threadIdx.x;
    if (t >= NTOK) return;

    float accR[E_], accN[E_];
#pragma unroll
    for (int e = 0; e < E_; ++e) { accR[e] = br[e]; accN[e] = bn[e]; }

    const float* xr = x + (size_t)t * D_;
    for (int d = 0; d < D_; ++d) {
        float xv = xr[d];
        const float* wr = Wr + (size_t)d * E_;   // uniform across threads -> s_load
        const float* wn = Wn + (size_t)d * E_;
#pragma unroll
        for (int e = 0; e < E_; ++e) {
            accR[e] = fmaf(xv, wr[e], accR[e]);
            accN[e] = fmaf(xv, wn[e], accN[e]);
        }
    }

    float noisy[E_];
#pragma unroll
    for (int e = 0; e < E_; ++e)
        noisy[e] = accR[e] + noise[(size_t)t * E_ + e] * softplus_f(accN[e]);

    // top-2 (first occurrence on ties, matching lax.top_k)
    int i0 = 0; float n0v = noisy[0];
#pragma unroll
    for (int e = 1; e < E_; ++e) if (noisy[e] > n0v) { n0v = noisy[e]; i0 = e; }
    int i1 = -1; float n1v = -3.4e38f;
#pragma unroll
    for (int e = 0; e < E_; ++e)
        if (e != i0 && noisy[e] > n1v) { n1v = noisy[e]; i1 = e; }

    // softmax over the two selected logits
    float g0 = fast_rcp(1.0f + __expf(n1v - n0v));
    topidx[t]  = make_int2(i0, i1);
    topgate[t] = make_float2(g0, 1.0f - g0);
}

// ------------------------------------------------- deterministic list build
__global__ void k_build_lists(const int2* __restrict__ topidx,
                              const float2* __restrict__ topgate,
                              int* __restrict__ tlist, float* __restrict__ tglist,
                              int* __restrict__ counts) {
    const int e = blockIdx.x;
    const int lane = threadIdx.x;        // 32 threads
    int cnt = 0;
    for (int base = 0; base < NTOK; base += 32) {
        int t = base + lane;
        int2 ti = topidx[t];
        float2 tg = topgate[t];
        bool sel = (ti.x == e) || (ti.y == e);
        float gv = (ti.x == e) ? tg.x : tg.y;
#if __has_builtin(__builtin_amdgcn_ballot_w32)
        unsigned mask = __builtin_amdgcn_ballot_w32(sel);
#else
        unsigned mask = (unsigned)__ballot(sel);
#endif
        int pos = cnt + __popc(mask & ((1u << lane) - 1u));
        if (sel) {
            tlist[e * NTOK + pos]  = t;
            tglist[e * NTOK + pos] = gv;
        }
        cnt += __popc(mask);
    }
    if (lane == 0) counts[e] = cnt;
}

// --------------------------------------------- stage 1: h = silu(xWg)*(xWu)
// tile: 128 tokens x 64 H-cols, K loop over D in chunks of 32
__global__ __launch_bounds__(256)
void k_expert_up(const __bf16* __restrict__ xb,
                 const float* __restrict__ Wg, const float* __restrict__ Wu,
                 const int* __restrict__ tlist,     // null => identity
                 const int* __restrict__ countPtr,  // null => NTOK
                 __bf16* __restrict__ hbuf) {
    const int count = countPtr ? *countPtr : NTOK;
    const int m0 = blockIdx.y * 128;
    if (m0 >= count) return;                 // uniform exit: EXEC stays all-1
    const int nbase = blockIdx.x * 64;

    __shared__ __align__(16) __bf16 sA [128 * 32];
    __shared__ __align__(16) __bf16 sBg[ 64 * 32];   // [n][k]
    __shared__ __align__(16) __bf16 sBu[ 64 * 32];

    const int tid  = threadIdx.x;
    const int lane = tid & 31;
    const int wave = tid >> 5;
    const int wm = wave & 3;       // 4 waves along M
    const int wn = wave >> 2;      // 2 waves along N

    // A-tile loading role: 2 threads per row, 32B each
    const int arow  = tid >> 1;
    const int ahalf = tid & 1;
    int r = m0 + arow;
    int tok = tlist ? tlist[r < count ? r : 0] : r;
    const __bf16* asrc_row = xb + (size_t)tok * D_;

    // B-tile loading role
    const int bn_ = tid & 63;
    const int bkb = (tid >> 6) * 8;

    v8f accg[2][2], accu[2][2];
#pragma unroll
    for (int i = 0; i < 2; ++i)
#pragma unroll
        for (int j = 0; j < 2; ++j) {
            accg[i][j] = (v8f){0, 0, 0, 0, 0, 0, 0, 0};
            accu[i][j] = (v8f){0, 0, 0, 0, 0, 0, 0, 0};
        }

    for (int k0 = 0; k0 < D_; k0 += 32) {
        __syncthreads();
        // A tile: bf16, async global->LDS (32B per thread)
        copy32_to_lds(asrc_row + k0 + ahalf * 16, sA + arow * 32 + ahalf * 16);
        {   // B tiles: f32 load (L2-resident), packed cvt, transposed [n][k]
            const float* g = Wg + (size_t)(k0 + bkb) * H_ + nbase + bn_;
            const float* u = Wu + (size_t)(k0 + bkb) * H_ + nbase + bn_;
            if (k0 + 32 < D_) {              // prefetch next chunk into near caches
                __builtin_prefetch(g + (size_t)32 * H_, 0, 3);
                __builtin_prefetch(u + (size_t)32 * H_, 0, 3);
            }
#pragma unroll
            for (int i = 0; i < 8; i += 2) {
                float ga = g[(size_t)i * H_], gb = g[(size_t)(i + 1) * H_];
                float ua = u[(size_t)i * H_], ub = u[(size_t)(i + 1) * H_];
                *(v2bf*)(sBg + bn_ * 32 + bkb + i) = (v2bf){(__bf16)ga, (__bf16)gb};
                *(v2bf*)(sBu + bn_ * 32 + bkb + i) = (v2bf){(__bf16)ua, (__bf16)ub};
            }
        }
        wait_lds_copy();
        __syncthreads();

        v16bf af[2];
#pragma unroll
        for (int fm = 0; fm < 2; ++fm) {
            int row = wm * 32 + fm * 16 + (lane & 15);
            int off = row * 32 + (lane >> 4) * 8;
            v8bf* p = (v8bf*)&af[fm];
            p[0] = *(const v8bf*)(sA + off);        // K 0..7   (or 8..15)
            p[1] = *(const v8bf*)(sA + off + 16);   // K 16..23 (or 24..31)
        }
        v16bf bg[2], bu[2];
#pragma unroll
        for (int fn = 0; fn < 2; ++fn) {
            int col = wn * 32 + fn * 16 + (lane & 15);
            int off = col * 32 + (lane >> 4) * 8;   // lane band selects K 0..7,16..23 or 8..15,24..31
            v8bf* pg = (v8bf*)&bg[fn];
            pg[0] = *(const v8bf*)(sBg + off);
            pg[1] = *(const v8bf*)(sBg + off + 16);
            v8bf* pu = (v8bf*)&bu[fn];
            pu[0] = *(const v8bf*)(sBu + off);
            pu[1] = *(const v8bf*)(sBu + off + 16);
        }
#pragma unroll
        for (int fm = 0; fm < 2; ++fm)
#pragma unroll
            for (int fn = 0; fn < 2; ++fn) {
                accg[fm][fn] = __builtin_amdgcn_wmma_f32_16x16x32_bf16(
                    false, af[fm], false, bg[fn], (short)0, accg[fm][fn], false, false);
                accu[fm][fn] = __builtin_amdgcn_wmma_f32_16x16x32_bf16(
                    false, af[fm], false, bu[fn], (short)0, accu[fm][fn], false, false);
            }
    }

    // epilogue: h = silu(g)*u via fast sigmoid, store bf16 rows
#pragma unroll
    for (int fm = 0; fm < 2; ++fm)
#pragma unroll
        for (int fn = 0; fn < 2; ++fn)
#pragma unroll
            for (int j = 0; j < 8; ++j) {
                int mrow = wm * 32 + fm * 16 + (lane >> 4) * 8 + j;
                int hcol = nbase + wn * 32 + fn * 16 + (lane & 15);
                float gv = accg[fm][fn][j];
                float uv = accu[fm][fn][j];
                float h  = gv * fast_rcp(1.0f + __expf(-gv)) * uv;
                hbuf[(size_t)(m0 + mrow) * H_ + hcol] = (__bf16)h;
            }
}

// --------------------------------------------- stage 2: y = h·Wd, scatter
// tile: 128 rows x 128 D-cols, K loop over H in chunks of 32
__global__ __launch_bounds__(256)
void k_expert_down(const __bf16* __restrict__ hbuf,
                   const float* __restrict__ Wd,      // [H_, D_]
                   const int* __restrict__ tlist,     // null => identity
                   const float* __restrict__ tglist,  // null => gate 1.0
                   const int* __restrict__ countPtr,  // null => NTOK
                   float* __restrict__ out,
                   int accumulate) {
    const int count = countPtr ? *countPtr : NTOK;
    const int m0 = blockIdx.y * 128;
    if (m0 >= count) return;
    const int nbase = blockIdx.x * 128;

    __shared__ __align__(16) __bf16 sA[128 * 32];
    __shared__ __align__(16) __bf16 sB[128 * 32];    // [n][k]

    const int tid  = threadIdx.x;
    const int lane = tid & 31;
    const int wave = tid >> 5;
    const int wm = wave & 3;
    const int wn = wave >> 2;

    const int arow  = tid >> 1;
    const int ahalf = tid & 1;
    const __bf16* asrc_row = hbuf + (size_t)(m0 + arow) * H_;

    const int bn_ = tid & 127;
    const int bkb = (tid >> 7) * 16;

    v8f acc[2][4];
#pragma unroll
    for (int i = 0; i < 2; ++i)
#pragma unroll
        for (int j = 0; j < 4; ++j) acc[i][j] = (v8f){0, 0, 0, 0, 0, 0, 0, 0};

    for (int k0 = 0; k0 < H_; k0 += 32) {
        __syncthreads();
        copy32_to_lds(asrc_row + k0 + ahalf * 16, sA + arow * 32 + ahalf * 16);
        {
            const float* w = Wd + (size_t)(k0 + bkb) * D_ + nbase + bn_;
            if (k0 + 32 < H_) __builtin_prefetch(w + (size_t)32 * D_, 0, 3);
#pragma unroll
            for (int i = 0; i < 16; i += 2) {
                float a = w[(size_t)i * D_], b = w[(size_t)(i + 1) * D_];
                *(v2bf*)(sB + bn_ * 32 + bkb + i) = (v2bf){(__bf16)a, (__bf16)b};
            }
        }
        wait_lds_copy();
        __syncthreads();

        v16bf af[2];
#pragma unroll
        for (int fm = 0; fm < 2; ++fm) {
            int row = wm * 32 + fm * 16 + (lane & 15);
            int off = row * 32 + (lane >> 4) * 8;
            v8bf* p = (v8bf*)&af[fm];
            p[0] = *(const v8bf*)(sA + off);
            p[1] = *(const v8bf*)(sA + off + 16);
        }
        v16bf bfr[4];
#pragma unroll
        for (int fn = 0; fn < 4; ++fn) {
            int col = wn * 64 + fn * 16 + (lane & 15);
            int off = col * 32 + (lane >> 4) * 8;
            v8bf* p = (v8bf*)&bfr[fn];
            p[0] = *(const v8bf*)(sB + off);
            p[1] = *(const v8bf*)(sB + off + 16);
        }
#pragma unroll
        for (int fm = 0; fm < 2; ++fm)
#pragma unroll
            for (int fn = 0; fn < 4; ++fn)
                acc[fm][fn] = __builtin_amdgcn_wmma_f32_16x16x32_bf16(
                    false, af[fm], false, bfr[fn], (short)0, acc[fm][fn], false, false);
    }

    // epilogue: out[tok] (+)= gate * y  — one writer per token per kernel
#pragma unroll
    for (int fm = 0; fm < 2; ++fm)
#pragma unroll
        for (int j = 0; j < 8; ++j) {
            int mrow = wm * 32 + fm * 16 + (lane >> 4) * 8 + j;
            int r = m0 + mrow;
            if (r < count) {
                int tok = tlist ? tlist[r] : r;
                float gate = tglist ? tglist[r] : 1.0f;
                float* orow = out + (size_t)tok * D_ + nbase + wn * 64 + (lane & 15);
#pragma unroll
                for (int fn = 0; fn < 4; ++fn) {
                    float v = gate * acc[fm][fn][j];
                    if (accumulate) orow[fn * 16] += v;
                    else            orow[fn * 16]  = v;
                }
            }
        }
}

// ---------------------------------------------------------------- launch
extern "C" void kernel_launch(void* const* d_in, const int* in_sizes, int n_in,
                              void* d_out, int out_size, void* d_ws, size_t ws_size,
                              hipStream_t stream) {
    const float* x     = (const float*)d_in[0];
    const float* noise = (const float*)d_in[1];
    const float* Wr    = (const float*)d_in[2];
    const float* br    = (const float*)d_in[3];
    const float* Wn    = (const float*)d_in[4];
    const float* bn    = (const float*)d_in[5];
    const float* Wg    = (const float*)d_in[6];
    const float* Wu    = (const float*)d_in[7];
    const float* Wd    = (const float*)d_in[8];
    const float* Sg    = (const float*)d_in[9];
    const float* Su    = (const float*)d_in[10];
    const float* Sd    = (const float*)d_in[11];
    float* out = (float*)d_out;

    char* ws = (char*)d_ws;
    auto alloc = [&](size_t bytes) {
        char* p = ws;
        ws += (bytes + 255) & ~(size_t)255;
        return p;
    };
    __bf16*  xb      = (__bf16*) alloc((size_t)NTOK * D_ * 2);
    __bf16*  hbuf    = (__bf16*) alloc((size_t)HROWS * H_ * 2);
    int2*    topidx  = (int2*)   alloc((size_t)NTOK * sizeof(int2));
    float2*  topgate = (float2*) alloc((size_t)NTOK * sizeof(float2));
    int*     tlist   = (int*)    alloc((size_t)E_ * NTOK * 4);
    float*   tglist  = (float*)  alloc((size_t)E_ * NTOK * 4);
    int*     counts  = (int*)    alloc((size_t)E_ * 4);

    k_cvt_bf16<<<(NTOK * D_ / 4 + 255) / 256, 256, 0, stream>>>(x, xb, NTOK * D_);
    k_router<<<(NTOK + 255) / 256, 256, 0, stream>>>(x, noise, Wr, br, Wn, bn,
                                                     topidx, topgate);
    k_build_lists<<<E_, 32, 0, stream>>>(topidx, topgate, tlist, tglist, counts);

    dim3 g1(H_ / 64, NTOK / 128);   // (64, 32)
    dim3 g2(D_ / 128, NTOK / 128);  // (8, 32)

    // shared expert initializes out
    k_expert_up  <<<g1, 256, 0, stream>>>(xb, Sg, Su, nullptr, nullptr, hbuf);
    k_expert_down<<<g2, 256, 0, stream>>>(hbuf, Sd, nullptr, nullptr, nullptr, out, 0);

    // routed experts, sequential on stream (deterministic accumulation)
    for (int e = 0; e < E_; ++e) {
        k_expert_up  <<<g1, 256, 0, stream>>>(xb,
                                              Wg + (size_t)e * D_ * H_,
                                              Wu + (size_t)e * D_ * H_,
                                              tlist + e * NTOK, counts + e, hbuf);
        k_expert_down<<<g2, 256, 0, stream>>>(hbuf,
                                              Wd + (size_t)e * H_ * D_,
                                              tlist + e * NTOK, tglist + e * NTOK,
                                              counts + e, out, 1);
    }
}
